// LinearCrossAttention_56126632624912
// MI455X (gfx1250) — matrix operand order repaired and checked by reading, not compiled, not run on hardware
//
#include <hip/hip_runtime.h>
#include <hip/hip_bf16.h>
#include <math.h>

typedef __attribute__((ext_vector_type(16))) __bf16       bf16x16;
typedef __attribute__((ext_vector_type(8)))  float        f32x8;

constexpr int BATCH = 8;
constexpr int DIM   = 512;   // channels == HIDDEN
constexpr int NTOK  = 4096;
constexpr int HEADS = 8;
constexpr int DHEAD = 64;

// ---------------------------------------------------------------------------
// Packed bf16-pair helpers. All LDS tiles hold 32-bit words = (K, K+1) bf16
// pairs arranged so every WMMA fragment is exactly two aligned ds_load_b128.
// ---------------------------------------------------------------------------
__device__ __forceinline__ unsigned pack_bf16(float lo, float hi) {
  union { __bf16 h[2]; unsigned u; } p;
  p.h[0] = (__bf16)lo;
  p.h[1] = (__bf16)hi;
  return p.u;
}

template <typename T> __device__ __forceinline__ float4 load4f(const T* p);
template <> __device__ __forceinline__ float4 load4f<float>(const float* p) {
  return *(const float4*)p;
}
template <> __device__ __forceinline__ float4 load4f<__bf16>(const __bf16* p) {
  union { uint2 u; __bf16 h[4]; } x;
  x.u = *(const uint2*)p;
  return make_float4((float)x.h[0], (float)x.h[1], (float)x.h[2], (float)x.h[3]);
}

__device__ __forceinline__ bf16x16 two_quads(uint4 lo, uint4 hi) {
  union { uint4 q[2]; bf16x16 v; } u;
  u.q[0] = lo; u.q[1] = hi;
  return u.v;
}

// A 16x32 bf16 (MxK): lane m = lane&15, half = lane>>4.
// Lane's words: row base + half*8 + {0..7}  ->  two ds_load_b128.
// lda2 must be a multiple of 4 words; k2ofs in multiples of 4.
__device__ __forceinline__ bf16x16 frag_a_pk(const unsigned* As2, int lda2,
                                             int row0, int k2ofs, int lane) {
  const int half = (lane >> 4) & 1;
  const int m = lane & 15;
  const uint4* q = (const uint4*)(As2 + (size_t)(row0 + m) * lda2 + k2ofs + (half << 3));
  return two_quads(q[0], q[1]);
}

// Fragment-major B tile: word (k2 in 0..15, n) stored at
//   (n>>4)*BSTRIDE + L*8 + (L>>3)*4 + (k2&7),   L = ((k2>>3)<<4)|(n&15)
// so lane L of tile (n>>4) reads its 8 words as two ds_load_b128.
constexpr int BSTRIDE = 272;   // words per 16-column tile (16B-aligned)
__device__ __forceinline__ int bfrag_addr(int k2, int n) {
  int L = ((k2 >> 3) << 4) | (n & 15);
  return (n >> 4) * BSTRIDE + L * 8 + ((L >> 3) << 2) + (k2 & 7);
}
__device__ __forceinline__ bf16x16 frag_b_pk(const unsigned* BsF, int tile, int lane) {
  const uint4* q = (const uint4*)(BsF + tile * BSTRIDE + lane * 8 + ((lane >> 3) << 2));
  return two_quads(q[0], q[1]);
}

__device__ __forceinline__ f32x8 wmma_bf16(bf16x16 a, bf16x16 b, f32x8 c) {
  return __builtin_amdgcn_wmma_f32_16x16x32_bf16(
      false, a, false, b, (short)0, c, false, false);
}

// Async 16B global -> LDS copy (CDNA5, ASYNCcnt-tracked). ldsOff is the raw
// LDS byte offset (flat LDS address truncates to addr[31:0] per ISA 10.2).
__device__ __forceinline__ void async_copy_b128(unsigned ldsOff, const void* gaddr) {
  asm volatile("global_load_async_to_lds_b128 %0, %1, off"
               :: "v"(ldsOff), "v"(gaddr) : "memory");
}
__device__ __forceinline__ void wait_asynccnt0() {
  asm volatile("s_wait_asynccnt 0x0" ::: "memory");
}

// ---------------------------------------------------------------------------
// Y[b,o,n] = sum_c W[o,c] * X[b,c,n] (+bias).  Block: 64(o) x 128(n) tile,
// 128 threads = 4 waves; each wave owns 16 o-rows x 8 WMMA n-tiles.
// ---------------------------------------------------------------------------
template <typename XT, bool ADD_BIAS, bool OUT_BF16>
__global__ void __launch_bounds__(128)
gemm512_kernel(const float* __restrict__ W, const XT* __restrict__ X,
               const float* __restrict__ bias, void* __restrict__ Yv) {
  __shared__ __align__(16) unsigned As2[64][20];        // o x k2 (16 pairs, pad->20)
  __shared__ __align__(16) unsigned BsF[8 * BSTRIDE];   // 8 fragment-major n-tiles
  const int t = threadIdx.x, wv = t >> 5, lane = t & 31;
  const int n0 = blockIdx.x * 128;
  const int o0 = blockIdx.y * 64;
  const int bb = blockIdx.z;
  const size_t xbase = (size_t)bb * DIM * NTOK;

  f32x8 acc[8] = {};
  for (int kc = 0; kc < DIM; kc += 32) {
#pragma unroll
    for (int i = 0; i < 2; ++i) {          // W tile: 64 rows x 16 words, uint4 stores
      int idx = t + i * 128;               // (r, p4): 64 x 4 groups
      int r = idx >> 2, p4 = (idx & 3) << 2;
      const float* wp = W + (size_t)(o0 + r) * DIM + kc + (p4 << 1);
      float4 w0 = *(const float4*)(wp);
      float4 w1 = *(const float4*)(wp + 4);
      *(uint4*)(&As2[r][p4]) = make_uint4(pack_bf16(w0.x, w0.y), pack_bf16(w0.z, w0.w),
                                          pack_bf16(w1.x, w1.y), pack_bf16(w1.z, w1.w));
    }
    const XT* xpf = nullptr;
#pragma unroll
    for (int i = 0; i < 4; ++i) {          // X tile: 16 k2 x 128 n words
      int idx = t + i * 128;               // (k2, c4): 16 x 32 groups
      int k2 = idx >> 5, c4 = (idx & 31) << 2;
      const XT* xp = X + xbase + (size_t)(kc + 2 * k2) * NTOK + n0 + c4;
      if (i == 0) xpf = xp;
      float4 a = load4f(xp);
      float4 b = load4f(xp + NTOK);
      BsF[bfrag_addr(k2, c4 + 0)] = pack_bf16(a.x, b.x);
      BsF[bfrag_addr(k2, c4 + 1)] = pack_bf16(a.y, b.y);
      BsF[bfrag_addr(k2, c4 + 2)] = pack_bf16(a.z, b.z);
      BsF[bfrag_addr(k2, c4 + 3)] = pack_bf16(a.w, b.w);
    }
    if (kc + 32 < DIM)                     // prefetch next k-step's X tile
      __builtin_prefetch((const void*)(xpf + 32 * NTOK), 0, 1);
    __syncthreads();
    bf16x16 a = frag_a_pk(&As2[0][0], 20, wv * 16, 0, lane);
#pragma unroll
    for (int j = 0; j < 8; ++j) {
      bf16x16 b = frag_b_pk(BsF, j, lane);
      acc[j] = wmma_bf16(a, b, acc[j]);
    }
    __syncthreads();
  }
  const int half = (lane >> 4) & 1, nl = lane & 15;
#pragma unroll
  for (int j = 0; j < 8; ++j) {
#pragma unroll
    for (int r = 0; r < 8; ++r) {
      int o = o0 + wv * 16 + half * 8 + r;
      int n = n0 + j * 16 + nl;
      float vv = acc[j][r];
      if (ADD_BIAS) vv += bias[o];
      size_t oi = (size_t)bb * DIM * NTOK + (size_t)o * NTOK + n;
      if (OUT_BF16) ((__bf16*)Yv)[oi] = (__bf16)vv;
      else          ((float*)Yv)[oi]  = vv;
    }
  }
}

// ---------------------------------------------------------------------------
// softmax over feature dim d (64) per (b,h,n), * DHEAD^-0.5 -> bf16
// ---------------------------------------------------------------------------
__global__ void __launch_bounds__(256)
softmax_q_kernel(const float* __restrict__ qp, __bf16* __restrict__ qb) {
  const int n = blockIdx.x * 256 + threadIdx.x;
  const int h = blockIdx.y, bb = blockIdx.z;
  const size_t base = ((size_t)bb * DIM + h * DHEAD) * NTOK + n;
  float v[DHEAD];
  float m = -1e30f;
#pragma unroll
  for (int d = 0; d < DHEAD; ++d) {
    v[d] = qp[base + (size_t)d * NTOK];
    m = fmaxf(m, v[d]);
  }
  float s = 0.f;
#pragma unroll
  for (int d = 0; d < DHEAD; ++d) { v[d] = __expf(v[d] - m); s += v[d]; }
  const float inv = 0.125f / s;   // SCALE = 64^-0.5
#pragma unroll
  for (int d = 0; d < DHEAD; ++d)
    qb[base + (size_t)d * NTOK] = (__bf16)(v[d] * inv);
}

// ---------------------------------------------------------------------------
// softmax over tokens n (4096) per channel row -> bf16
// ---------------------------------------------------------------------------
__global__ void __launch_bounds__(256)
softmax_k_kernel(const float* __restrict__ kp, __bf16* __restrict__ kb) {
  __shared__ float red[256];
  const int t = threadIdx.x;
  const size_t base = (size_t)blockIdx.x * NTOK;  // row = b*DIM + c
  const float4* kp4 = (const float4*)(kp + base);
  float4 v[4];
  float m = -1e30f;
#pragma unroll
  for (int j = 0; j < 4; ++j) {
    v[j] = kp4[t + j * 256];
    m = fmaxf(m, fmaxf(fmaxf(v[j].x, v[j].y), fmaxf(v[j].z, v[j].w)));
  }
  red[t] = m; __syncthreads();
  for (int s = 128; s > 0; s >>= 1) {
    if (t < s) red[t] = fmaxf(red[t], red[t + s]);
    __syncthreads();
  }
  m = red[0]; __syncthreads();
  float sum = 0.f;
#pragma unroll
  for (int j = 0; j < 4; ++j) {
    v[j].x = __expf(v[j].x - m); v[j].y = __expf(v[j].y - m);
    v[j].z = __expf(v[j].z - m); v[j].w = __expf(v[j].w - m);
    sum += v[j].x + v[j].y + v[j].z + v[j].w;
  }
  red[t] = sum; __syncthreads();
  for (int s = 128; s > 0; s >>= 1) {
    if (t < s) red[t] += red[t + s];
    __syncthreads();
  }
  const float inv = 1.f / red[0];
  unsigned* kbu = (unsigned*)(kb + base);
#pragma unroll
  for (int j = 0; j < 4; ++j) {
    kbu[(t + j * 256) * 2 + 0] = pack_bf16(v[j].x * inv, v[j].y * inv);
    kbu[(t + j * 256) * 2 + 1] = pack_bf16(v[j].z * inv, v[j].w * inv);
  }
}

__global__ void zero_f32(float* p, int nElem) {
  int i = blockIdx.x * blockDim.x + threadIdx.x;
  if (i < nElem) p[i] = 0.f;
}

// ---------------------------------------------------------------------------
// ctx[b,h,d,e] = sum_n kh[d,n]*vh[e,n]; split-K(8) + f32 atomics.
// Tiles are pure 16B copies of pre-packed bf16 pairs -> async global->LDS
// (GLOBAL_LOAD_ASYNC_TO_LDS_B128, ASYNCcnt) with no VGPR round-trip.
// ---------------------------------------------------------------------------
__global__ void __launch_bounds__(128)
context_kernel(const __bf16* __restrict__ kb, const __bf16* __restrict__ vb,
               float* __restrict__ ctx) {
  __shared__ __align__(16) unsigned Ak2[64][20];        // d x n2
  __shared__ __align__(16) unsigned BvF[4 * BSTRIDE];   // 4 fragment-major e-tiles
  const int t = threadIdx.x, wv = t >> 5, lane = t & 31;
  const int chunk = blockIdx.x, h = blockIdx.y, bb = blockIdx.z;
  const size_t hbase = ((size_t)bb * DIM + h * DHEAD) * NTOK;
  const uint4* kbq = (const uint4*)(kb + hbase);
  const uint4* vbq = (const uint4*)(vb + hbase);
  // Per-thread tile slots (r = row, qd = quad-of-pairs), fixed across k-steps.
  const int r0 = t >> 2,           qd0 = t & 3;
  const int r1 = (t + 128) >> 2,   qd1 = (t + 128) & 3;
  const unsigned ldsA0 = (unsigned)(size_t)&Ak2[r0][qd0 << 2];
  const unsigned ldsA1 = (unsigned)(size_t)&Ak2[r1][qd1 << 2];
  const unsigned ldsB0 = (unsigned)(size_t)&BvF[bfrag_addr(qd0 << 2, r0)];
  const unsigned ldsB1 = (unsigned)(size_t)&BvF[bfrag_addr(qd1 << 2, r1)];

  f32x8 acc[4] = {};
  for (int nc = chunk * 512; nc < chunk * 512 + 512; nc += 32) {
    const size_t g0 = (size_t)r0 * (NTOK / 8) + (nc >> 3) + qd0;
    const size_t g1 = (size_t)r1 * (NTOK / 8) + (nc >> 3) + qd1;
    async_copy_b128(ldsA0, kbq + g0);
    async_copy_b128(ldsB0, vbq + g0);
    async_copy_b128(ldsA1, kbq + g1);
    async_copy_b128(ldsB1, vbq + g1);
    wait_asynccnt0();
    __syncthreads();
    bf16x16 a = frag_a_pk(&Ak2[0][0], 20, wv * 16, 0, lane);
#pragma unroll
    for (int j = 0; j < 4; ++j) {
      bf16x16 b = frag_b_pk(BvF, j, lane);
      acc[j] = wmma_bf16(a, b, acc[j]);
    }
    __syncthreads();
  }
  const int half = (lane >> 4) & 1, nl = lane & 15;
  float* cbase = ctx + ((size_t)bb * HEADS + h) * DHEAD * DHEAD;
#pragma unroll
  for (int j = 0; j < 4; ++j)
#pragma unroll
    for (int r = 0; r < 8; ++r) {
      int d = wv * 16 + half * 8 + r;
      int e = j * 16 + nl;
      atomicAdd(&cbase[d * DHEAD + e], acc[j][r]);
    }
}

// ---------------------------------------------------------------------------
// ao[b,h,e,n] = sum_d ctx[d,e] * qh[d,n]  (M=e, K=d=64, N=n)
// ---------------------------------------------------------------------------
__global__ void __launch_bounds__(128)
attnout_kernel(const float* __restrict__ ctx, const __bf16* __restrict__ qb,
               __bf16* __restrict__ ao) {
  __shared__ __align__(16) unsigned As2[64][36];        // e x d2 (32 pairs, pad->36)
  __shared__ __align__(16) unsigned BsF[4 * BSTRIDE];   // 4 fragment-major n-tiles
  const int t = threadIdx.x, wv = t >> 5, lane = t & 31;
  const int n0 = blockIdx.x * 64, h = blockIdx.y, bb = blockIdx.z;
  const float* cb = ctx + ((size_t)bb * HEADS + h) * DHEAD * DHEAD;
#pragma unroll
  for (int i = 0; i < 4; ++i) {            // ctx^T: 32 d2 x 16 e-quads
    int idx = t + i * 128;
    int d2 = idx >> 4, e4 = (idx & 15) << 2;
    float4 c0 = *(const float4*)(cb + (2 * d2) * DHEAD + e4);
    float4 c1 = *(const float4*)(cb + (2 * d2 + 1) * DHEAD + e4);
    As2[e4 + 0][d2] = pack_bf16(c0.x, c1.x);
    As2[e4 + 1][d2] = pack_bf16(c0.y, c1.y);
    As2[e4 + 2][d2] = pack_bf16(c0.z, c1.z);
    As2[e4 + 3][d2] = pack_bf16(c0.w, c1.w);
  }
  const size_t hbase = ((size_t)bb * DIM + h * DHEAD) * NTOK;
  f32x8 acc[4] = {};
  for (int kd = 0; kd < DHEAD; kd += 32) {
    __syncthreads();
#pragma unroll
    for (int i = 0; i < 2; ++i) {          // qh tile: 16 d2 x 16 n-quads
      int idx = t + i * 128;
      int d2 = idx >> 4, c4 = (idx & 15) << 2;
      const __bf16* qp = qb + hbase + (size_t)(kd + 2 * d2) * NTOK + n0 + c4;
      float4 a = load4f(qp);
      float4 b = load4f(qp + NTOK);
      BsF[bfrag_addr(d2, c4 + 0)] = pack_bf16(a.x, b.x);
      BsF[bfrag_addr(d2, c4 + 1)] = pack_bf16(a.y, b.y);
      BsF[bfrag_addr(d2, c4 + 2)] = pack_bf16(a.z, b.z);
      BsF[bfrag_addr(d2, c4 + 3)] = pack_bf16(a.w, b.w);
    }
    __syncthreads();
    bf16x16 a = frag_a_pk(&As2[0][0], 36, wv * 16, kd >> 1, lane);
#pragma unroll
    for (int j = 0; j < 4; ++j) {
      bf16x16 b = frag_b_pk(BsF, j, lane);
      acc[j] = wmma_bf16(a, b, acc[j]);
    }
  }
  const int half = (lane >> 4) & 1, nl = lane & 15;
#pragma unroll
  for (int j = 0; j < 4; ++j)
#pragma unroll
    for (int r = 0; r < 8; ++r) {
      int e = wv * 16 + half * 8 + r;
      int n = n0 + j * 16 + nl;
      ao[hbase + (size_t)e * NTOK + n] = (__bf16)acc[j][r];
    }
}

// ---------------------------------------------------------------------------
// per-(b,n) L2-normalize over 512 channels, * g[c] * sqrt(512); 2 cols/thread
// ---------------------------------------------------------------------------
__global__ void __launch_bounds__(256)
rmsnorm_kernel(const float* __restrict__ x, const float* __restrict__ g,
               float* __restrict__ out) {
  const int n = (blockIdx.x * 256 + threadIdx.x) * 2;
  const int bb = blockIdx.y;
  const size_t base = (size_t)bb * DIM * NTOK + n;
  float s0 = 0.f, s1 = 0.f;
#pragma unroll 8
  for (int c = 0; c < DIM; ++c) {
    float2 v = *(const float2*)(x + base + (size_t)c * NTOK);
    s0 += v.x * v.x;
    s1 += v.y * v.y;
  }
  const float r0 = 22.62741699796952f / fmaxf(sqrtf(s0), 1e-12f);
  const float r1 = 22.62741699796952f / fmaxf(sqrtf(s1), 1e-12f);
#pragma unroll 8
  for (int c = 0; c < DIM; ++c) {
    float2 v = *(const float2*)(x + base + (size_t)c * NTOK);
    float gc = g[c];
    float2 o = make_float2(v.x * r0 * gc, v.y * r1 * gc);
    *(float2*)(out + base + (size_t)c * NTOK) = o;
  }
}

// ---------------------------------------------------------------------------

extern "C" void kernel_launch(void* const* d_in, const int* in_sizes, int n_in,
                              void* d_out, int out_size, void* d_ws, size_t ws_size,
                              hipStream_t stream) {
  const float* q  = (const float*)d_in[0];
  const float* k  = (const float*)d_in[1];
  const float* v  = (const float*)d_in[2];
  const float* Wq = (const float*)d_in[3];
  const float* Wk = (const float*)d_in[4];
  const float* Wv = (const float*)d_in[5];
  const float* Wo = (const float*)d_in[6];
  const float* bo = (const float*)d_in[7];
  const float* g  = (const float*)d_in[8];
  float* out = (float*)d_out;

  char* w = (char*)d_ws;
  const size_t F32SZ  = (size_t)BATCH * DIM * NTOK * sizeof(float);  // 64 MiB
  const size_t BF16SZ = F32SZ / 2;
  float*  qp  = (float*)(w);                                   // q proj (fp32)
  float*  kp  = (float*)(w + F32SZ);                           // k proj (fp32)
  __bf16* qb  = (__bf16*)(w + 2 * F32SZ);                      // softmax(q)*scale
  __bf16* kb  = (__bf16*)(w + 2 * F32SZ + BF16SZ);             // softmax(k)
  __bf16* vb  = (__bf16*)(w + 2 * F32SZ + 2 * BF16SZ);         // v proj (bf16)
  __bf16* ao  = (__bf16*)(w + 2 * F32SZ + 3 * BF16SZ);         // attention out
  float*  ctx = (float*)(w + 2 * F32SZ + 4 * BF16SZ);          // [b,h,64,64]
  float*  tmp = qp;                                            // reuse for pre-norm

  dim3 gGemm(NTOK / 128, DIM / 64, BATCH);
  gemm512_kernel<float, false, false><<<gGemm, 128, 0, stream>>>(Wq, q, nullptr, qp);
  gemm512_kernel<float, false, false><<<gGemm, 128, 0, stream>>>(Wk, k, nullptr, kp);
  gemm512_kernel<float, false, true ><<<gGemm, 128, 0, stream>>>(Wv, v, nullptr, vb);

  softmax_q_kernel<<<dim3(NTOK / 256, HEADS, BATCH), 256, 0, stream>>>(qp, qb);
  softmax_k_kernel<<<BATCH * DIM, 256, 0, stream>>>(kp, kb);

  const int ctxElems = BATCH * HEADS * DHEAD * DHEAD;
  zero_f32<<<(ctxElems + 255) / 256, 256, 0, stream>>>(ctx, ctxElems);
  context_kernel<<<dim3(8, HEADS, BATCH), 128, 0, stream>>>(kb, vb, ctx);
  attnout_kernel<<<dim3(NTOK / 64, HEADS, BATCH), 128, 0, stream>>>(ctx, qb, ao);

  gemm512_kernel<__bf16, true, false><<<gGemm, 128, 0, stream>>>(Wo, ao, bo, tmp);
  rmsnorm_kernel<<<dim3(NTOK / 512, BATCH), 256, 0, stream>>>(tmp, g, out);
}
